// MultiHeadAttention_73014444032677
// MI455X (gfx1250) — matrix operand order repaired and checked by reading, not compiled, not run on hardware
//
#include <hip/hip_runtime.h>
#include <hip/hip_bf16.h>

// ---------------------------------------------------------------- types ----
typedef __bf16 bf16;
typedef __attribute__((ext_vector_type(16))) __bf16 v16bf;
typedef __attribute__((ext_vector_type(8)))  float  v8f;

union AFrag { v16bf v; unsigned u[8]; };
union BFrag { v16bf v; unsigned u[8]; };
union CFrag { v8f   v; float    f[8]; };

// ---------------------------------------------------------------- consts ---
#define D_MODEL   1024
#define NUM_HEADS 16
#define DEPTH     64
#define BATCH     4
#define SEQ       2048
#define ROWS      (BATCH * SEQ)        // 8192
#define NKT       (SEQ / 64)           // 32 key tiles
#define NKG       (D_MODEL / 32)       // 32 GEMM k-steps

#define AS_STR 40   // 128x32 A tile LDS stride (halves) = 16 DW row + 4 DW pad
#define BT_STR 40   // 64x32 B tile stride (same pad)
#define K_STR  72   // 64x64 attention tile stride = 32 DW row + 4 DW pad
#define P_STR  72   // per-wave P scratch stride

// A-operand k order used here (contiguous k = 16*hh + 2*j2, same as B reads): element pair j2 -> K index
__device__ __forceinline__ int a_klocal(int j2, int hh) {
    return 2 * j2 + hh * 16;
}

#define LDS_OFF(p) ((unsigned)(unsigned long long)(const void*)(p))

// --------------------------------------------------------- TDM (CDNA5) -----
#if defined(__HIP_DEVICE_COMPILE__) &&                                         \
    __has_builtin(__builtin_amdgcn_tensor_load_to_lds) &&                      \
    __has_builtin(__builtin_amdgcn_s_wait_tensorcnt)
#define USE_TDM 1
#else
#define USE_TDM 0
#endif

#if USE_TDM
typedef unsigned int u32x4 __attribute__((ext_vector_type(4)));
typedef int          i32x8 __attribute__((ext_vector_type(8)));
typedef int          i32x4 __attribute__((ext_vector_type(4)));

// 2-D bf16 tile load with descriptor-controlled LDS row padding.
// D# bitfields per CDNA5 ISA 8.3/8.4. 6-arg toolchain builtin:
//   (u32x4 g0, i32x8 g1, i32x4 g2, i32x4 g3, i32x8 extra, i32 cpol)
// pad_int: interval = 2<<code DWORDs ; pad_amt: (code+1) DWORDs inserted.
__device__ __forceinline__ void tdm_load_tile(unsigned lds_off,
                                              const void* gptr,
                                              unsigned td0, unsigned td1,
                                              unsigned tile0, unsigned tile1,
                                              unsigned long long stride0,
                                              int pad_int, int pad_amt) {
    unsigned long long ga = (unsigned long long)gptr;
    u32x4 g0;
    g0[0] = 1u;                                    // count=1, user descriptor
    g0[1] = lds_off;                               // lds_addr (bytes)
    g0[2] = (unsigned)ga;                          // global_addr[31:0]
    g0[3] = ((unsigned)(ga >> 32) & 0x01FFFFFFu)   // global_addr[56:32]
            | (2u << 30);                          // type = 2 (image)
    i32x8 g1;
    g1[0] = (1 << 16)            // data_size = 2 bytes (bf16)
          | (1 << 20)            // pad_enable
          | (pad_int << 22)      // pad_interval
          | (pad_amt << 25);     // pad_amount
    g1[1] = (int)(td0 << 16);                       // tensor_dim0[15:0]
    g1[2] = (int)((td0 >> 16) | (td1 << 16));       // dim0[31:16] | dim1[15:0]
    g1[3] = (int)((td1 >> 16) | (tile0 << 16));     // dim1[31:16] | tile_dim0
    g1[4] = (int)tile1;                             // tile_dim1 (tile_dim2=0)
    g1[5] = (int)(stride0 & 0xFFFFFFFFu);           // dim0_stride[31:0]
    g1[6] = (int)((stride0 >> 32) & 0xFFFFu);       // dim0_stride[47:32]
    g1[7] = 0;
    i32x4 z4 = {0, 0, 0, 0};                        // groups 2/3 unused (2-D)
    i32x8 z8 = {0, 0, 0, 0, 0, 0, 0, 0};            // extra words unused
    __builtin_amdgcn_tensor_load_to_lds(g0, g1, z4, z4, z8, 0);
}
#endif

// ---------------------------------------------------------------- casts ----
__global__ void cast_f32_to_bf16(const float* __restrict__ in,
                                 bf16* __restrict__ out, int n4) {
    int i = blockIdx.x * 256 + threadIdx.x;
    if (i < n4) {
        float4 f = ((const float4*)in)[i];
        out[4 * i + 0] = (bf16)f.x;
        out[4 * i + 1] = (bf16)f.y;
        out[4 * i + 2] = (bf16)f.z;
        out[4 * i + 3] = (bf16)f.w;
    }
}

// in: [Z][R][C] fp32 (row-major)  ->  out: [Z][C][R] bf16  (32x32 LDS tiles)
// grid = (C/32, R/32, Z), block = (32, 8)
__global__ void cast_transpose_bf16(const float* __restrict__ in,
                                    bf16* __restrict__ out, int R, int C) {
    __shared__ float t[32][33];
    const size_t plane = (size_t)R * C;
    const float* src = in + (size_t)blockIdx.z * plane;
    bf16*        dst = out + (size_t)blockIdx.z * plane;
    const int c0 = blockIdx.x * 32, r0 = blockIdx.y * 32;
#pragma unroll
    for (int i = threadIdx.y; i < 32; i += 8)
        t[i][threadIdx.x] = src[(size_t)(r0 + i) * C + c0 + threadIdx.x];
    __syncthreads();
#pragma unroll
    for (int i = threadIdx.y; i < 32; i += 8)
        dst[(size_t)(c0 + i) * R + r0 + threadIdx.x] = (bf16)t[threadIdx.x][i];
}

// ------------------------------------------------------------- QKV GEMM ----
// grid.x = ROWS/128, grid.y = 3*NUM_HEADS ; block = 256 (8 waves)
// weights pre-transposed: wt* = [H, DEPTH, D_MODEL] bf16
// Q : [B*H, S, DEPTH] bf16, (acc+bias)*1/sqrt(64) folded in
// K : [B*H, S, DEPTH] bf16
// V : [B*H, DEPTH, S] bf16  (pre-transposed so attention can TDM-load V^T)
__global__ __launch_bounds__(256, 2)
void qkv_kernel(const bf16* __restrict__ xb,
                const bf16* __restrict__ wqt, const bf16* __restrict__ wkt,
                const bf16* __restrict__ wvt,
                const float* __restrict__ bq, const float* __restrict__ bk,
                const float* __restrict__ bv,
                bf16* __restrict__ qo, bf16* __restrict__ ko,
                bf16* __restrict__ vo) {
    __shared__ bf16 Abuf[2][128 * AS_STR];
    __shared__ bf16 Bbuf[2][64 * BT_STR];

    const int ph   = blockIdx.y;
    const int proj = ph / NUM_HEADS;
    const int h    = ph % NUM_HEADS;
    const bf16*  wt   = (proj == 0) ? wqt : (proj == 1) ? wkt : wvt;
    const float* bias = (proj == 0) ? bq : (proj == 1) ? bk : bv;
    const float  scl  = (proj == 0) ? 0.125f : 1.0f;

    const int row0 = blockIdx.x * 128;
    const int tid  = threadIdx.x;
    const int lane = tid & 31, wave = tid >> 5;
    const int m = lane & 15, hh = lane >> 4;

    CFrag acc[4];
#pragma unroll
    for (int nt = 0; nt < 4; nt++)
#pragma unroll
        for (int r = 0; r < 8; r++) acc[nt].f[r] = 0.0f;

    const bf16* wht = wt + h * (DEPTH * D_MODEL);   // [64][1024]

#if USE_TDM
    if (wave == 0) {
        tdm_load_tile(LDS_OFF(&Abuf[0][0]), &xb[(long)row0 * D_MODEL],
                      D_MODEL, ROWS, 32, 128, D_MODEL, 3, 3);
        tdm_load_tile(LDS_OFF(&Bbuf[0][0]), wht,
                      D_MODEL, DEPTH, 32, 64, D_MODEL, 3, 3);
    }
#endif

    for (int it = 0; it < NKG; ++it) {
        const int cur = it & 1;
        const int k0  = it * 32;
#if USE_TDM
        if (wave == 0) {
            if (it + 1 < NKG) {
                tdm_load_tile(LDS_OFF(&Abuf[cur ^ 1][0]),
                              &xb[(long)row0 * D_MODEL + k0 + 32],
                              D_MODEL, ROWS, 32, 128, D_MODEL, 3, 3);
                tdm_load_tile(LDS_OFF(&Bbuf[cur ^ 1][0]), &wht[k0 + 32],
                              D_MODEL, DEPTH, 32, 64, D_MODEL, 3, 3);
                __builtin_amdgcn_s_wait_tensorcnt((short)2);
            } else {
                __builtin_amdgcn_s_wait_tensorcnt((short)0);
            }
        }
        __syncthreads();  // buf[cur] ready
#else
        __syncthreads();
#pragma unroll
        for (int c = tid; c < 512; c += 256) {
            int r = c >> 2, cg = c & 3;
            *(uint4*)&Abuf[0][r * AS_STR + cg * 8] =
                *(const uint4*)&xb[(row0 + r) * D_MODEL + k0 + cg * 8];
        }
        {
            int n = tid & 63, kk0 = (tid >> 6) * 8;
            *(uint4*)&Bbuf[0][n * BT_STR + kk0] =
                *(const uint4*)&wht[n * D_MODEL + k0 + kk0];
        }
        __syncthreads();
#endif
        const bf16* Al = Abuf[USE_TDM ? cur : 0];
        const bf16* Bl = Bbuf[USE_TDM ? cur : 0];

        AFrag a;
#pragma unroll
        for (int j2 = 0; j2 < 8; j2++)
            a.u[j2] = *(const unsigned*)&Al[(wave * 16 + m) * AS_STR +
                                            a_klocal(j2, hh)];
#pragma unroll
        for (int nt = 0; nt < 4; nt++) {
            BFrag b;
#pragma unroll
            for (int j2 = 0; j2 < 8; j2++)
                b.u[j2] = *(const unsigned*)&Bl[(nt * 16 + m) * BT_STR +
                                                hh * 16 + 2 * j2];
            acc[nt].v = __builtin_amdgcn_wmma_f32_16x16x32_bf16(
                false, a.v, false, b.v, (short)0, acc[nt].v, false, false);
        }
#if USE_TDM
        __syncthreads();  // reads of buf[cur] done before TDM refills it
#endif
    }

    const int b  = row0 / SEQ;
    const int s0 = row0 - b * SEQ + wave * 16;
    if (proj == 2) {
        // V^T: [B*H, DEPTH, SEQ]
        const long tb = (long)(b * NUM_HEADS + h) * DEPTH;
#pragma unroll
        for (int nt = 0; nt < 4; nt++) {
            int   e   = nt * 16 + m;
            float bia = bias[h * DEPTH + e];
#pragma unroll
            for (int r = 0; r < 8; r++) {
                int srow = s0 + r + 8 * hh;
                vo[(tb + e) * (long)SEQ + srow] = (bf16)(acc[nt].f[r] + bia);
            }
        }
    } else {
        bf16* out = (proj == 0) ? qo : ko;
        const long obas = (long)(b * NUM_HEADS + h) * SEQ;
#pragma unroll
        for (int nt = 0; nt < 4; nt++) {
            int   e   = nt * 16 + m;
            float bia = bias[h * DEPTH + e];
#pragma unroll
            for (int r = 0; r < 8; r++) {
                int srow = s0 + r + 8 * hh;
                out[(obas + srow) * DEPTH + e] =
                    (bf16)((acc[nt].f[r] + bia) * scl);
            }
        }
    }
}

// ------------------------------------------------------ flash attention ----
// grid.x = SEQ/128 query blocks, grid.y = B*H ; block = 256 (8 waves)
// K/V^T tiles DMA'd into double-buffered LDS by the Tensor Data Mover,
// overlapping the next tile's DMA with this tile's 16 WMMAs.
__global__ __launch_bounds__(256, 2)
void attn_kernel(const bf16* __restrict__ qb, const bf16* __restrict__ kb,
                 const bf16* __restrict__ vt, bf16* __restrict__ multi) {
    __shared__ bf16 Kbuf[2][64 * K_STR];
    __shared__ bf16 Vbuf[2][64 * K_STR];
    __shared__ bf16 Pl[8 * 16 * P_STR];

    const int bh = blockIdx.y;
    const int bb = bh / NUM_HEADS, h = bh % NUM_HEADS;
    const int q0 = blockIdx.x * 128;
    const int tid = threadIdx.x, lane = tid & 31, wave = tid >> 5;
    const int m = lane & 15, hh = lane >> 4;
    const long base = (long)bh * SEQ;
    const long vtb  = (long)bh * DEPTH * SEQ;

    // Q fragments straight from global ([s,64] row-major: pairs contiguous)
    AFrag qf[2];
    {
        const bf16* qrow = &qb[(base + q0 + wave * 16 + m) * DEPTH];
#pragma unroll
        for (int ks = 0; ks < 2; ks++)
#pragma unroll
            for (int j2 = 0; j2 < 8; j2++)
                qf[ks].u[j2] =
                    *(const unsigned*)&qrow[ks * 32 + a_klocal(j2, hh)];
    }

    CFrag acc[4];
    float mi[8], li[8];
#pragma unroll
    for (int nt = 0; nt < 4; nt++)
#pragma unroll
        for (int r = 0; r < 8; r++) acc[nt].f[r] = 0.0f;
#pragma unroll
    for (int r = 0; r < 8; r++) { mi[r] = -3.0e38f; li[r] = 0.0f; }

#if USE_TDM
    if (wave == 0) {  // one wave drives the DMA engine
        tdm_load_tile(LDS_OFF(&Kbuf[0][0]), &kb[base * DEPTH],
                      DEPTH, SEQ, 64, 64, DEPTH, 4, 3);
        tdm_load_tile(LDS_OFF(&Vbuf[0][0]), &vt[vtb],
                      SEQ, DEPTH, 64, 64, SEQ, 4, 3);
    }
#endif

    for (int it = 0; it < NKT; ++it) {
        const int cur = it & 1;
        const int kt  = it * 64;
        const bf16* Kl = Kbuf[cur];
        const bf16* Vl = Vbuf[cur];

#if USE_TDM
        if (wave == 0) {
            if (it + 1 < NKT) {
                const int nkt = kt + 64;
                tdm_load_tile(LDS_OFF(&Kbuf[cur ^ 1][0]),
                              &kb[(base + nkt) * DEPTH],
                              DEPTH, SEQ, 64, 64, DEPTH, 4, 3);
                tdm_load_tile(LDS_OFF(&Vbuf[cur ^ 1][0]), &vt[vtb + nkt],
                              SEQ, DEPTH, 64, 64, SEQ, 4, 3);
                // in-order TDM: <=2 outstanding means tile `it` has landed
                __builtin_amdgcn_s_wait_tensorcnt((short)2);
            } else {
                __builtin_amdgcn_s_wait_tensorcnt((short)0);
            }
        }
        __syncthreads();  // buf[cur] visible to all waves
#else
        __syncthreads();  // previous tile's readers done
#pragma unroll
        for (int c = tid; c < 512; c += 256) {
            int r = c >> 3, cg = c & 7;
            *(uint4*)&Kbuf[cur][r * K_STR + cg * 8] =
                *(const uint4*)&kb[(base + kt + r) * DEPTH + cg * 8];
            *(uint4*)&Vbuf[cur][r * K_STR + cg * 8] =
                *(const uint4*)&vt[vtb + (long)r * SEQ + kt + cg * 8];
        }
        __syncthreads();
#endif

        // scores = Q * K^T   (per wave: 16 rows x 64 keys)
        CFrag sc[4];
#pragma unroll
        for (int nt = 0; nt < 4; nt++) {
#pragma unroll
            for (int r = 0; r < 8; r++) sc[nt].f[r] = 0.0f;
#pragma unroll
            for (int ks = 0; ks < 2; ks++) {
                BFrag b;
#pragma unroll
                for (int j2 = 0; j2 < 8; j2++)
                    b.u[j2] = *(const unsigned*)&Kl[(nt * 16 + m) * K_STR +
                                                    ks * 32 + hh * 16 + 2 * j2];
                sc[nt].v = __builtin_amdgcn_wmma_f32_16x16x32_bf16(
                    false, qf[ks].v, false, b.v, (short)0, sc[nt].v, false,
                    false);
            }
        }

        // online softmax (rows r+8*hh on this lane-half; cols across 16 lanes)
        float mnew[8], corr[8], rs[8];
#pragma unroll
        for (int r = 0; r < 8; r++) {
            float v = fmaxf(fmaxf(sc[0].f[r], sc[1].f[r]),
                            fmaxf(sc[2].f[r], sc[3].f[r]));
#pragma unroll
            for (int off = 8; off >= 1; off >>= 1)
                v = fmaxf(v, __shfl_xor(v, off, 32));
            mnew[r] = fmaxf(mi[r], v);
            corr[r] = __expf(mi[r] - mnew[r]);
        }
#pragma unroll
        for (int r = 0; r < 8; r++) {
            float s = 0.0f;
#pragma unroll
            for (int nt = 0; nt < 4; nt++) {
                sc[nt].f[r] = __expf(sc[nt].f[r] - mnew[r]);
                s += sc[nt].f[r];
            }
#pragma unroll
            for (int off = 8; off >= 1; off >>= 1) s += __shfl_xor(s, off, 32);
            rs[r] = s;
        }
#pragma unroll
        for (int r = 0; r < 8; r++) {
            li[r] = li[r] * corr[r] + rs[r];
            mi[r] = mnew[r];
        }
#pragma unroll
        for (int nt = 0; nt < 4; nt++)
#pragma unroll
            for (int r = 0; r < 8; r++) acc[nt].f[r] *= corr[r];

        // C-layout -> A-layout transpose of P via per-wave LDS scratch
        bf16* pw = &Pl[wave * 16 * P_STR];
#pragma unroll
        for (int nt = 0; nt < 4; nt++)
#pragma unroll
            for (int r = 0; r < 8; r++)
                pw[(r + 8 * hh) * P_STR + nt * 16 + m] = (bf16)sc[nt].f[r];
        asm volatile("s_wait_dscnt 0x0" ::: "memory");  // same-wave DS order

        // acc += P * V
#pragma unroll
        for (int ks = 0; ks < 2; ks++) {
            AFrag a;
#pragma unroll
            for (int j2 = 0; j2 < 8; j2++)
                a.u[j2] = *(const unsigned*)&pw[m * P_STR + ks * 32 +
                                                a_klocal(j2, hh)];
#pragma unroll
            for (int nt = 0; nt < 4; nt++) {
                BFrag b;
#pragma unroll
                for (int j2 = 0; j2 < 8; j2++)
                    b.u[j2] = *(const unsigned*)&Vl[(nt * 16 + m) * K_STR +
                                                    ks * 32 + hh * 16 + 2 * j2];
                acc[nt].v = __builtin_amdgcn_wmma_f32_16x16x32_bf16(
                    false, a.v, false, b.v, (short)0, acc[nt].v, false, false);
            }
        }

#if USE_TDM
        __syncthreads();  // all reads of buf[cur] done before it is re-filled
#endif
    }

    // epilogue: heads concatenated -> multi[b, s, h*64+e]
    const int s0 = q0 + wave * 16;
#pragma unroll
    for (int nt = 0; nt < 4; nt++) {
        int e = nt * 16 + m;
#pragma unroll
        for (int r = 0; r < 8; r++) {
            int s = s0 + r + 8 * hh;
            multi[((long)(bb * SEQ + s)) * D_MODEL + h * DEPTH + e] =
                (bf16)(acc[nt].f[r] / li[r]);
        }
    }
}

// ----------------------------------------------------- output projection ---
// grid.x = ROWS/128, grid.y = D_MODEL/64 ; block = 256 (8 waves), fp32 out
// wot = Wo^T : [D_MODEL out_cols][D_MODEL k] bf16
__global__ __launch_bounds__(256, 2)
void oproj_kernel(const bf16* __restrict__ multi, const bf16* __restrict__ wot,
                  const float* __restrict__ bo, float* __restrict__ out) {
    __shared__ bf16 Abuf[2][128 * AS_STR];
    __shared__ bf16 Bbuf[2][64 * BT_STR];

    const int row0 = blockIdx.x * 128;
    const int n0   = blockIdx.y * 64;
    const int tid  = threadIdx.x;
    const int lane = tid & 31, wave = tid >> 5;
    const int m = lane & 15, hh = lane >> 4;

    CFrag acc[4];
#pragma unroll
    for (int nt = 0; nt < 4; nt++)
#pragma unroll
        for (int r = 0; r < 8; r++) acc[nt].f[r] = 0.0f;

    const bf16* wn = &wot[(long)n0 * D_MODEL];   // 64 output cols x 1024 k

#if USE_TDM
    if (wave == 0) {
        tdm_load_tile(LDS_OFF(&Abuf[0][0]), &multi[(long)row0 * D_MODEL],
                      D_MODEL, ROWS, 32, 128, D_MODEL, 3, 3);
        tdm_load_tile(LDS_OFF(&Bbuf[0][0]), wn,
                      D_MODEL, DEPTH, 32, 64, D_MODEL, 3, 3);
    }
#endif

    for (int it = 0; it < NKG; ++it) {
        const int cur = it & 1;
        const int k0  = it * 32;
#if USE_TDM
        if (wave == 0) {
            if (it + 1 < NKG) {
                tdm_load_tile(LDS_OFF(&Abuf[cur ^ 1][0]),
                              &multi[(long)row0 * D_MODEL + k0 + 32],
                              D_MODEL, ROWS, 32, 128, D_MODEL, 3, 3);
                tdm_load_tile(LDS_OFF(&Bbuf[cur ^ 1][0]), &wn[k0 + 32],
                              D_MODEL, DEPTH, 32, 64, D_MODEL, 3, 3);
                __builtin_amdgcn_s_wait_tensorcnt((short)2);
            } else {
                __builtin_amdgcn_s_wait_tensorcnt((short)0);
            }
        }
        __syncthreads();
#else
        __syncthreads();
#pragma unroll
        for (int c = tid; c < 512; c += 256) {
            int r = c >> 2, cg = c & 3;
            *(uint4*)&Abuf[0][r * AS_STR + cg * 8] =
                *(const uint4*)&multi[(row0 + r) * D_MODEL + k0 + cg * 8];
        }
        {
            int n = tid & 63, kk0 = (tid >> 6) * 8;
            *(uint4*)&Bbuf[0][n * BT_STR + kk0] =
                *(const uint4*)&wn[(long)n * D_MODEL + k0 + kk0];
        }
        __syncthreads();
#endif
        const bf16* Al = Abuf[USE_TDM ? cur : 0];
        const bf16* Bl = Bbuf[USE_TDM ? cur : 0];

        AFrag a;
#pragma unroll
        for (int j2 = 0; j2 < 8; j2++)
            a.u[j2] = *(const unsigned*)&Al[(wave * 16 + m) * AS_STR +
                                            a_klocal(j2, hh)];
#pragma unroll
        for (int nt = 0; nt < 4; nt++) {
            BFrag b;
#pragma unroll
            for (int j2 = 0; j2 < 8; j2++)
                b.u[j2] = *(const unsigned*)&Bl[(nt * 16 + m) * BT_STR +
                                                hh * 16 + 2 * j2];
            acc[nt].v = __builtin_amdgcn_wmma_f32_16x16x32_bf16(
                false, a.v, false, b.v, (short)0, acc[nt].v, false, false);
        }
#if USE_TDM
        __syncthreads();
#endif
    }

#pragma unroll
    for (int nt = 0; nt < 4; nt++) {
        int e = n0 + nt * 16 + m;
#pragma unroll
        for (int r = 0; r < 8; r++) {
            int row = row0 + wave * 16 + r + 8 * hh;
            out[(long)row * D_MODEL + e] = acc[nt].f[r] + bo[e];
        }
    }
}

// ---------------------------------------------------------------- launch ---
extern "C" void kernel_launch(void* const* d_in, const int* in_sizes, int n_in,
                              void* d_out, int out_size, void* d_ws,
                              size_t ws_size, hipStream_t stream) {
    const float* x  = (const float*)d_in[0];
    const float* Wq = (const float*)d_in[1];
    const float* bq = (const float*)d_in[2];
    const float* Wk = (const float*)d_in[3];
    const float* bk = (const float*)d_in[4];
    const float* Wv = (const float*)d_in[5];
    const float* bv = (const float*)d_in[6];
    const float* Wo = (const float*)d_in[7];
    const float* bo = (const float*)d_in[8];
    float* out = (float*)d_out;

    const size_t N_X = (size_t)ROWS * D_MODEL;                 // 8388608
    const size_t N_W = (size_t)NUM_HEADS * D_MODEL * DEPTH;    // 1048576
    bf16* xb  = (bf16*)d_ws;
    bf16* wqt = xb + N_X;    // [H, DEPTH, D_MODEL]
    bf16* wkt = wqt + N_W;
    bf16* wvt = wkt + N_W;
    bf16* wot = wvt + N_W;   // [D_MODEL, D_MODEL] = Wo^T
    bf16* qbf = wot + (size_t)D_MODEL * D_MODEL;
    bf16* kbf = qbf + N_X;
    bf16* vbf = kbf + N_X;   // stored as [B*H, DEPTH, SEQ] (V^T)
    bf16* mbf = vbf + N_X;

    // fp32 -> bf16 casts (weights transposed on the way)
    cast_f32_to_bf16<<<(int)(N_X / 4 + 255) / 256, 256, 0, stream>>>(
        x, xb, (int)(N_X / 4));
    dim3 tb(32, 8);
    cast_transpose_bf16<<<dim3(DEPTH / 32, D_MODEL / 32, NUM_HEADS), tb, 0,
                          stream>>>(Wq, wqt, D_MODEL, DEPTH);
    cast_transpose_bf16<<<dim3(DEPTH / 32, D_MODEL / 32, NUM_HEADS), tb, 0,
                          stream>>>(Wk, wkt, D_MODEL, DEPTH);
    cast_transpose_bf16<<<dim3(DEPTH / 32, D_MODEL / 32, NUM_HEADS), tb, 0,
                          stream>>>(Wv, wvt, D_MODEL, DEPTH);
    cast_transpose_bf16<<<dim3(D_MODEL / 32, D_MODEL / 32, 1), tb, 0,
                          stream>>>(Wo, wot, D_MODEL, D_MODEL);

    qkv_kernel<<<dim3(ROWS / 128, 3 * NUM_HEADS), 256, 0, stream>>>(
        xb, wqt, wkt, wvt, bq, bk, bv, qbf, kbf, vbf);

    attn_kernel<<<dim3(SEQ / 128, BATCH * NUM_HEADS), 256, 0, stream>>>(
        qbf, kbf, vbf, mbf);

    oproj_kernel<<<dim3(ROWS / 128, D_MODEL / 64), 256, 0, stream>>>(
        mbf, wot, bo, out);
}
